// MultiHeadAttention_40767829574222
// MI455X (gfx1250) — matrix operand order repaired and checked by reading, not compiled, not run on hardware
//
#include <hip/hip_runtime.h>

typedef __attribute__((ext_vector_type(16))) __bf16 v16bf;
typedef __attribute__((ext_vector_type(8)))  __bf16 v8bf;
typedef __attribute__((ext_vector_type(4)))  __bf16 v4bf;
typedef __attribute__((ext_vector_type(8)))  float  v8f;

namespace {
constexpr int kB = 4, kS = 2048, kD = 1024, kH = 16, kDK = 64;
constexpr int kM = kB * kS;  // 8192 rows of activations
}

__device__ __forceinline__ v16bf cat8(v8bf lo, v8bf hi) {
  return __builtin_shufflevector(lo, hi, 0,1,2,3,4,5,6,7,8,9,10,11,12,13,14,15);
}

// Vectorized fp32 -> bf16 conversion: 4 elements/thread (float4 in, 8B out).
__global__ void cvt_f32_bf16x4(const float* __restrict__ s, __bf16* __restrict__ d, int n4) {
  int i = blockIdx.x * blockDim.x + threadIdx.x;
  if (i < n4) {
    float4 f = *(const float4*)(s + 4 * (size_t)i);
    v4bf o;
    o[0] = (__bf16)f.x; o[1] = (__bf16)f.y; o[2] = (__bf16)f.z; o[3] = (__bf16)f.w;
    *(v4bf*)(d + 4 * (size_t)i) = o;
  }
}

// C[m,n] = X[m,:] . W[n,:] + bias[n]   (torch Linear, W stored [N,K] row-major)
// X is always bf16 [m,k]. Epilogue MODE:
//   0: out bf16 scattered to q/k head layout [b,h,s,dk]
//   1: out bf16 scattered to v-transposed layout [b,h,dk,s]
//   2: out fp32 row-major [m,n] (final projection to d_out)
template <int MODE>
__global__ __launch_bounds__(256)
void gemm_wmma(const __bf16* __restrict__ X, const __bf16* __restrict__ Wb,
               const float* __restrict__ bias, void* __restrict__ Yv) {
  const int lane = threadIdx.x & 31;
  const int wave = threadIdx.x >> 5;
  const int r  = lane & 15;
  const int hi = lane >> 4;
  const int m0 = blockIdx.x * 128 + (wave & 3) * 32;   // 2 M-tiles of 16
  const int n0 = blockIdx.y * 128 + (wave >> 2) * 64;  // 4 N-tiles of 16

  v8f acc[2][4];
#pragma unroll
  for (int i = 0; i < 2; ++i)
#pragma unroll
    for (int j = 0; j < 4; ++j) acc[i][j] = {};

  for (int kb = 0; kb < kD; kb += 32) {
    v16bf a[2];
#pragma unroll
    for (int i = 0; i < 2; ++i) {
      const __bf16* ap = X + (size_t)(m0 + i * 16 + r) * kD + kb;
      a[i] = *(const v16bf*)(ap + 16 * hi);
    }
    v16bf bf[4];
#pragma unroll
    for (int j = 0; j < 4; ++j)
      bf[j] = *(const v16bf*)(Wb + (size_t)(n0 + j * 16 + r) * kD + kb + 16 * hi);
#pragma unroll
    for (int i = 0; i < 2; ++i)
#pragma unroll
      for (int j = 0; j < 4; ++j)
        acc[i][j] = __builtin_amdgcn_wmma_f32_16x16x32_bf16(
            false, a[i], false, bf[j], (short)0, acc[i][j], false, false);
  }

#pragma unroll
  for (int i = 0; i < 2; ++i)
#pragma unroll
    for (int j = 0; j < 4; ++j)
#pragma unroll
      for (int v = 0; v < 8; ++v) {
        const int m = m0 + i * 16 + 8 * hi + v;
        const int n = n0 + j * 16 + r;
        const float val = acc[i][j][v] + bias[n];
        if constexpr (MODE == 2) {
          ((float*)Yv)[(size_t)m * kD + n] = val;
        } else {
          const int b = m >> 11, s = m & 2047;   // m = b*2048 + s
          const int h = n >> 6,  dk = n & 63;    // n = h*64 + dk
          if constexpr (MODE == 0)
            ((__bf16*)Yv)[(((size_t)b * kH + h) * kS + s) * kDK + dk] = (__bf16)val;
          else
            ((__bf16*)Yv)[(((size_t)b * kH + h) * kDK + dk) * kS + s] = (__bf16)val;
        }
      }
}

// Flash attention: one wave handles one 16-row q tile over all 2048 keys,
// 64 keys per iteration. q,k: [b,h,s,dk] bf16; vt: [b,h,dk,s] bf16;
// out: [b,s,h*64+dk] bf16.
__global__ __launch_bounds__(128)
void attn_wmma(const __bf16* __restrict__ q, const __bf16* __restrict__ k,
               const __bf16* __restrict__ vt, __bf16* __restrict__ attn) {
  __shared__ __bf16 lds_p[4][16 * 64];
  const int lane = threadIdx.x & 31;
  const int wave = threadIdx.x >> 5;
  const int r  = lane & 15;
  const int hi = lane >> 4;
  const int qt = blockIdx.x * 4 + wave;   // global q-tile id, 0..8191
  const int bh = qt >> 7;                 // / (S/16)
  const int qbase = (qt & 127) << 4;
  const int b = bh >> 4, h = bh & 15;

  // Q tile A-fragments (held for whole loop)
  const __bf16* qrow = q + ((size_t)bh * kS + qbase + r) * kDK;
  const v16bf qa0 = *(const v16bf*)(qrow + 16 * hi);
  const v16bf qa1 = *(const v16bf*)(qrow + 32 + 16 * hi);

  v8f oacc[4];
#pragma unroll
  for (int t = 0; t < 4; ++t) oacc[t] = {};
  float mrow[8], lrow[8];
#pragma unroll
  for (int v = 0; v < 8; ++v) { mrow[v] = -3.0e38f; lrow[v] = 0.0f; }
  __bf16* lp = lds_p[wave];

  for (int kb = 0; kb < kS; kb += 64) {
    // scores: four 16x16 tiles (keys kb+16g .. kb+16g+15), contract dk=64
    v8f sc[4];
#pragma unroll
    for (int g = 0; g < 4; ++g) {
      sc[g] = {};
      const __bf16* kg = k + ((size_t)bh * kS + kb + g * 16 + r) * kDK;
      v16bf kf0 = *(const v16bf*)(kg + 16 * hi);
      v16bf kf1 = *(const v16bf*)(kg + 32 + 16 * hi);
      sc[g] = __builtin_amdgcn_wmma_f32_16x16x32_bf16(false, qa0, false, kf0, (short)0, sc[g], false, false);
      sc[g] = __builtin_amdgcn_wmma_f32_16x16x32_bf16(false, qa1, false, kf1, (short)0, sc[g], false, false);
    }
    float p[4][8];
#pragma unroll
    for (int v = 0; v < 8; ++v) {
      float a0 = sc[0][v] * 0.125f, a1 = sc[1][v] * 0.125f;  // 1/sqrt(64)
      float a2 = sc[2][v] * 0.125f, a3 = sc[3][v] * 0.125f;
      float x = fmaxf(fmaxf(a0, a1), fmaxf(a2, a3));
      x = fmaxf(x, __shfl_xor(x, 1, 32));
      x = fmaxf(x, __shfl_xor(x, 2, 32));
      x = fmaxf(x, __shfl_xor(x, 4, 32));
      x = fmaxf(x, __shfl_xor(x, 8, 32));           // row max within 16-lane half
      const float nm = fmaxf(mrow[v], x);
      const float alpha = __expf(mrow[v] - nm);
      mrow[v] = nm;
      p[0][v] = __expf(a0 - nm);
      p[1][v] = __expf(a1 - nm);
      p[2][v] = __expf(a2 - nm);
      p[3][v] = __expf(a3 - nm);
      float rs = (p[0][v] + p[1][v]) + (p[2][v] + p[3][v]);
      rs += __shfl_xor(rs, 1, 32);
      rs += __shfl_xor(rs, 2, 32);
      rs += __shfl_xor(rs, 4, 32);
      rs += __shfl_xor(rs, 8, 32);
      lrow[v] = lrow[v] * alpha + rs;
#pragma unroll
      for (int t = 0; t < 4; ++t) oacc[t][v] *= alpha;
    }
    // re-swizzle P (C-layout) -> A-fragment layout via LDS (16x64 tile)
#pragma unroll
    for (int v = 0; v < 8; ++v) {
      const int m = v + 8 * hi;
#pragma unroll
      for (int g = 0; g < 4; ++g) lp[m * 64 + g * 16 + r] = (__bf16)p[g][v];
    }
    asm volatile("s_wait_dscnt 0" ::: "memory");
    const __bf16* pr = lp + r * 64;
    const v16bf pa0 = *(const v16bf*)(pr + 16 * hi);
    const v16bf pa1 = *(const v16bf*)(pr + 32 + 16 * hi);
    // O += P . V : B-fragments contiguous from transposed V, K=64 contraction
#pragma unroll
    for (int t = 0; t < 4; ++t) {
      const __bf16* vr = vt + ((size_t)bh * kDK + t * 16 + r) * kS + kb;
      v16bf vb0 = *(const v16bf*)(vr + 16 * hi);
      v16bf vb1 = *(const v16bf*)(vr + 32 + 16 * hi);
      oacc[t] = __builtin_amdgcn_wmma_f32_16x16x32_bf16(false, pa0, false, vb0, (short)0, oacc[t], false, false);
      oacc[t] = __builtin_amdgcn_wmma_f32_16x16x32_bf16(false, pa1, false, vb1, (short)0, oacc[t], false, false);
    }
  }
#pragma unroll
  for (int v = 0; v < 8; ++v) {
    const float inv = 1.0f / lrow[v];
    const int m = v + 8 * hi;
#pragma unroll
    for (int t = 0; t < 4; ++t)
      attn[((size_t)b * kS + qbase + m) * kD + h * kDK + t * 16 + r] = (__bf16)(oacc[t][v] * inv);
  }
}

extern "C" void kernel_launch(void* const* d_in, const int* in_sizes, int n_in,
                              void* d_out, int out_size, void* d_ws, size_t ws_size,
                              hipStream_t stream) {
  const float* Q  = (const float*)d_in[0];
  const float* K  = (const float*)d_in[1];
  const float* V  = (const float*)d_in[2];
  const float* Wq = (const float*)d_in[3];
  const float* bq = (const float*)d_in[4];
  const float* Wk = (const float*)d_in[5];
  const float* bk = (const float*)d_in[6];
  const float* Wv = (const float*)d_in[7];
  const float* bv = (const float*)d_in[8];
  const float* Wo = (const float*)d_in[9];
  const float* bo = (const float*)d_in[10];

  char* ws = (char*)d_ws;
  const size_t WSZ = (size_t)kD * kD * sizeof(__bf16);             // 2 MB per weight
  const size_t ASZ = (size_t)kM * kD * sizeof(__bf16);             // 16 MB per activation
  __bf16* Wq16 = (__bf16*)(ws + 0 * WSZ);
  __bf16* Wk16 = (__bf16*)(ws + 1 * WSZ);
  __bf16* Wv16 = (__bf16*)(ws + 2 * WSZ);
  __bf16* Wo16 = (__bf16*)(ws + 3 * WSZ);
  __bf16* Xq   = (__bf16*)(ws + 4 * WSZ);            // bf16 copy of Q input
  __bf16* Xk   = (__bf16*)(ws + 4 * WSZ + 1 * ASZ);  // bf16 copy of K input
  __bf16* Xv   = (__bf16*)(ws + 4 * WSZ + 2 * ASZ);  // bf16 copy of V input
  __bf16* qws  = (__bf16*)(ws + 4 * WSZ + 3 * ASZ);  // q [b,h,s,dk]
  __bf16* kws  = (__bf16*)(ws + 4 * WSZ + 4 * ASZ);  // k [b,h,s,dk]
  __bf16* vtws = (__bf16*)(ws + 4 * WSZ + 5 * ASZ);  // v [b,h,dk,s]
  // attn output aliases Xq: Xq's last reader (q-projection GEMM) is ordered
  // before attn_wmma on the stream, so reuse is deterministic every call.
  __bf16* aws  = Xq;

  const int nW4 = kD * kD / 4;       // weight elements / 4
  const int nA4 = kM * kD / 4;       // activation elements / 4
  cvt_f32_bf16x4<<<nW4 / 256, 256, 0, stream>>>(Wq, Wq16, nW4);
  cvt_f32_bf16x4<<<nW4 / 256, 256, 0, stream>>>(Wk, Wk16, nW4);
  cvt_f32_bf16x4<<<nW4 / 256, 256, 0, stream>>>(Wv, Wv16, nW4);
  cvt_f32_bf16x4<<<nW4 / 256, 256, 0, stream>>>(Wo, Wo16, nW4);
  cvt_f32_bf16x4<<<nA4 / 256, 256, 0, stream>>>(Q, Xq, nA4);
  cvt_f32_bf16x4<<<nA4 / 256, 256, 0, stream>>>(K, Xk, nA4);
  cvt_f32_bf16x4<<<nA4 / 256, 256, 0, stream>>>(V, Xv, nA4);

  dim3 gg(kM / 128, kD / 128), gb(256);
  gemm_wmma<0><<<gg, gb, 0, stream>>>(Xq, Wq16, bq, qws);
  gemm_wmma<0><<<gg, gb, 0, stream>>>(Xk, Wk16, bk, kws);
  gemm_wmma<1><<<gg, gb, 0, stream>>>(Xv, Wv16, bv, vtws);

  attn_wmma<<<(kB * kH * (kS / 16)) / 4, 128, 0, stream>>>(qws, kws, vtws, aws);

  gemm_wmma<2><<<gg, gb, 0, stream>>>(aws, Wo16, bo, d_out);
}
